// CapsNet_45612552683670
// MI455X (gfx1250) — matrix operand order repaired and checked by reading, not compiled, not run on hardware
//
#include <hip/hip_runtime.h>
#include <hip/hip_bf16.h>
#include <stdint.h>

typedef __attribute__((ext_vector_type(16))) _Float16 v16h;
typedef __attribute__((ext_vector_type(8)))  float    v8f;

#define THRESH 0.5f
#define DECAY  0.2f
#define TRACE_DECAY 0.51341712f   // exp(-1/1.5)

// ---------------- sizes ----------------
#define CONV_N   3276800          // [32,256,20,20]
#define PRIM_N   294912           // [32,1152,8]
#define DIG_N    5898240          // [32,1152,10,16]
#define KB       648              // k-blocks of 32 (K = 256*81 = 20736)
#define KSPLIT   4
#define KSTEPS   (KB / KSPLIT)    // 162
#define BPACK_N  (16*KB*512)      // 5,308,416  halves
#define APACK_N  (72*KB*512)      // 23,887,872 halves

// ---------------- workspace offsets (bytes) ----------------
#define OFF_CONV_MEM   0u
#define OFF_PRIM_MEM   (OFF_CONV_MEM + CONV_N*4u)
#define OFF_PRIM_SPK   (OFF_PRIM_MEM + PRIM_N*4u)
#define OFF_DIG_MEM    (OFF_PRIM_SPK + PRIM_N*4u)
#define OFF_DIG_SPK    (OFF_DIG_MEM + DIG_N*4u)
#define OFF_TRACE      (OFF_DIG_SPK + DIG_N*4u)
#define OFF_DIG2_MEM   (OFF_TRACE + DIG_N*4u)
#define OFF_DIG2_SPK   (OFF_DIG2_MEM + 5120u*4u)
#define OFF_OUT_MEM    (OFF_DIG2_SPK + 5120u*4u)
#define OFF_CONV_SPK   (OFF_OUT_MEM + 5120u*4u)            // f16
#define ZERO_END       (OFF_CONV_SPK + CONV_N*2u)          // zero-init region end
#define OFF_BIJ        ZERO_END
#define OFF_CONV_OUT   (OFF_BIJ + 11520u*4u)
#define OFF_BPACK      (OFF_CONV_OUT + CONV_N*4u)
#define OFF_APACK      (OFF_BPACK + BPACK_N*2u)
#define OFF_PRIM_PART  (OFF_APACK + (size_t)APACK_N*2u)    // 4 * PRIM_N f32
#define OFF_SJ         (OFF_PRIM_PART + (size_t)KSPLIT*PRIM_N*4u)

// ---------------- init ----------------
__global__ void k_zero(uint32_t* __restrict__ p, int n) {
  int i = blockIdx.x * blockDim.x + threadIdx.x;
  int stride = gridDim.x * blockDim.x;
  for (; i < n; i += stride) p[i] = 0u;
}

__global__ void k_init_bij(float* __restrict__ b) {
  int i = blockIdx.x * blockDim.x + threadIdx.x;
  if (i < 11520) b[i] = 1.0f / 1152.0f;
}

// Pack prim_w [co=256][ci=256][9][9] f32 into WMMA B-fragment order (f16).
// One thread per fragment row (nb,kb,lane): 16*648*32 threads, 32B vector store.
// Lane layout (B 32x16 f16): n = nb*16+(lane&15); h[i] -> K = kb*32 + (i<8 ? (lane>>4)*8+i : 16+(lane>>4)*8+(i-8))
__global__ void k_pack_b(const float* __restrict__ w, _Float16* __restrict__ Bp) {
  int p = blockIdx.x * blockDim.x + threadIdx.x;
  if (p >= 16 * KB * 32) return;
  int lane = p & 31;
  int kb   = (p >> 5) % KB;
  int nb   = p / (32 * KB);
  int n  = nb * 16 + (lane & 15);
  int K0 = kb * 32 + (lane >> 4) * 8;
  const float* wp = w + (size_t)n * 20736 + K0;
  v16h v;
#pragma unroll
  for (int i = 0; i < 16; i++) v[i] = (_Float16)wp[i < 8 ? i : i + 8];
  *(v16h*)(Bp + (size_t)p * 16) = v;
}

// conv1 (loop-invariant, computed once)
__global__ void k_conv1(const float* __restrict__ data, const float* __restrict__ w,
                        const float* __restrict__ bias, float* __restrict__ out) {
  int idx = blockIdx.x * blockDim.x + threadIdx.x;
  if (idx >= CONV_N) return;
  int ox = idx % 20;
  int oy = (idx / 20) % 20;
  int co = (idx / 400) & 255;
  int b  = idx / 102400;
  const float* dp = data + b * 784 + oy * 28 + ox;
  const float* wp = w + co * 81;
  float acc = bias[co];
#pragma unroll
  for (int ky = 0; ky < 9; ky++)
#pragma unroll
    for (int kx = 0; kx < 9; kx++)
      acc = fmaf(dp[ky * 28 + kx], wp[ky * 9 + kx], acc);
  out[idx] = fmaxf(acc, 0.0f);
}

// ---------------- per-timestep kernels ----------------
__global__ void k_conv_update(float* __restrict__ mem, _Float16* __restrict__ spike,
                              const float* __restrict__ conv_out) {
  int i = blockIdx.x * blockDim.x + threadIdx.x;
  if (i >= CONV_N) return;
  float sp = (float)spike[i];
  float m = (mem[i] - sp * THRESH) * DECAY + conv_out[i];
  mem[i] = m;
  spike[i] = (_Float16)(m > THRESH ? 1.0f : 0.0f);
}

// im2col of conv_spike into WMMA A-fragment order (f16).
// One thread per fragment row (mb,kb,lane): 72*648*32 threads, 32B vector store.
// A 16x32 f16 layout: m = mb*16+(lane&15); half = lane>>4;
//   h[i] -> K = kb*32 + (i<8 ? half*8+i : 16+half*8+(i-8))
__global__ void k_apack(const _Float16* __restrict__ spike, _Float16* __restrict__ Ap) {
  int p = blockIdx.x * blockDim.x + threadIdx.x;
  if (p >= 72 * KB * 32) return;
  int lane = p & 31;
  int kb   = (p >> 5) % KB;
  int mb   = p / (32 * KB);
  int half = lane >> 4;
  int m    = mb * 16 + (lane & 15);
  int b  = m / 36;  int s  = m - b * 36;
  int oy = s / 6;   int ox = s - oy * 6;
  const _Float16* sp = spike + (size_t)b * 102400 + (oy * 2) * 20 + ox * 2;
  v16h v;
#pragma unroll
  for (int g = 0; g < 2; ++g) {
    int K  = kb * 32 + half * 8 + g * 16;   // 8 consecutive K per group
    int ci = K / 81;
    int rem = K - ci * 81;
    int ky = rem / 9;
    int kx = rem - ky * 9;
#pragma unroll
    for (int i = 0; i < 8; ++i) {
      v[g * 8 + i] = sp[ci * 400 + ky * 20 + kx];
      if (++kx == 9) { kx = 0; if (++ky == 9) { ky = 0; ++ci; } }
    }
  }
  *(v16h*)(Ap + (size_t)p * 16) = v;
}

// WMMA implicit-GEMM, 32x32 register tile per wave (2x2 fragments, 4 wmma/k-step),
// split-K=4 for occupancy (1152 waves), deterministic partial buffers.
__global__ __launch_bounds__(256) void k_prim_gemm(const _Float16* __restrict__ Apack,
                                                   const _Float16* __restrict__ Bpack,
                                                   float* __restrict__ part) {
  const int lane = threadIdx.x & 31;
  const int wave = threadIdx.x >> 5;
  const int tile = blockIdx.x * 8 + wave;   // 0..1151
  const int ks   = tile & 3;                // split-K piece
  const int nbB  = (tile >> 2) & 7;         // 32-col block, 0..7
  const int mbB  = tile >> 5;               // 32-row block, 0..35
  const int kb0  = ks * KSTEPS;
  const _Float16* a0 = Apack + ((size_t)((2 * mbB    ) * KB + kb0) * 32 + lane) * 16;
  const _Float16* a1 = Apack + ((size_t)((2 * mbB + 1) * KB + kb0) * 32 + lane) * 16;
  const _Float16* b0 = Bpack + ((size_t)((2 * nbB    ) * KB + kb0) * 32 + lane) * 16;
  const _Float16* b1 = Bpack + ((size_t)((2 * nbB + 1) * KB + kb0) * 32 + lane) * 16;
  v8f c00 = {}, c01 = {}, c10 = {}, c11 = {};
#pragma unroll 2
  for (int kb = 0; kb < KSTEPS; ++kb) {
    v16h A0 = *(const v16h*)a0;
    v16h A1 = *(const v16h*)a1;
    v16h B0 = *(const v16h*)b0;
    v16h B1 = *(const v16h*)b1;
    c00 = __builtin_amdgcn_wmma_f32_16x16x32_f16(false, A0, false, B0, (short)0, c00, false, false);
    c01 = __builtin_amdgcn_wmma_f32_16x16x32_f16(false, A0, false, B1, (short)0, c01, false, false);
    c10 = __builtin_amdgcn_wmma_f32_16x16x32_f16(false, A1, false, B0, (short)0, c10, false, false);
    c11 = __builtin_amdgcn_wmma_f32_16x16x32_f16(false, A1, false, B1, (short)0, c11, false, false);
    __builtin_prefetch(a0 + 16 * 512, 0, 1);   // global_prefetch_b8, 16 k-steps ahead
    __builtin_prefetch(a1 + 16 * 512, 0, 1);
    __builtin_prefetch(b0 + 16 * 512, 0, 1);
    __builtin_prefetch(b1 + 16 * 512, 0, 1);
    a0 += 512; a1 += 512; b0 += 512; b1 += 512;
  }
  float* dst = part + (size_t)ks * PRIM_N;
  const int half = lane >> 4;
  const int nlo  = lane & 15;
#define STORE_C(CC, MB, NB)                                            \
  {                                                                    \
    const int n = (NB) * 16 + nlo;                                     \
    _Pragma("unroll")                                                  \
    for (int j = 0; j < 8; j++) {                                      \
      int m = (MB) * 16 + j + 8 * half;                                \
      int b_ = m / 36; int s = m - b_ * 36;                            \
      dst[b_ * 9216 + n * 36 + s] = CC[j];                             \
    }                                                                  \
  }
  STORE_C(c00, 2 * mbB,     2 * nbB)
  STORE_C(c01, 2 * mbB,     2 * nbB + 1)
  STORE_C(c10, 2 * mbB + 1, 2 * nbB)
  STORE_C(c11, 2 * mbB + 1, 2 * nbB + 1)
#undef STORE_C
}

// combine split-K partials + bias, then membrane/spike update
__global__ void k_prim_update(float* __restrict__ mem, float* __restrict__ spike,
                              const float* __restrict__ part,
                              const float* __restrict__ prim_b) {
  int i = blockIdx.x * blockDim.x + threadIdx.x;
  if (i >= PRIM_N) return;
  int n = (i % 9216) / 36;   // output channel (i = b*9216 + n*36 + s)
  float pout = part[i] + part[PRIM_N + i] + part[2 * PRIM_N + i] +
               part[3 * PRIM_N + i] + prim_b[n];
  float m = (mem[i] - spike[i] * THRESH) * DECAY + pout;
  mem[i] = m;
  spike[i] = m > THRESH ? 1.0f : 0.0f;
}

// Fused: u_hat = W@x + bias ; dig membrane/spike ; trace_u update
__global__ void k_dig_update(const float* __restrict__ W, const float* __restrict__ bias16,
                             const float* __restrict__ pspike, float* __restrict__ mem,
                             float* __restrict__ spike, float* __restrict__ trace) {
  int idx = blockIdx.x * blockDim.x + threadIdx.x;
  if (idx >= DIG_N) return;
  int o  = idx & 15;
  int t  = idx >> 4;
  int c  = t % 10;
  int t2 = t / 10;
  int r  = t2 % 1152;
  int b  = t2 / 1152;
  const float* wp = W + ((size_t)(r * 10 + c) * 16 + o) * 8;
  const float* xp = pspike + (b * 1152 + r) * 8;
  float u = bias16[o];
#pragma unroll
  for (int i = 0; i < 8; i++) u = fmaf(wp[i], xp[i], u);
  float m = (mem[idx] - spike[idx] * THRESH) * DECAY + u;
  mem[idx] = m;
  float s = m > THRESH ? 1.0f : 0.0f;
  spike[idx] = s;
  trace[idx] = (s > 0.5f) ? 1.0f : trace[idx] * TRACE_DECAY;
}

// s_j[b,c,o] = sum_r b_ij[r,c] * dig_spike[b,r,c,o] ; one block per (b,c)
__global__ void k_sj(const float* __restrict__ bij, const float* __restrict__ spike,
                     float* __restrict__ sj) {
  int bc = blockIdx.x;          // 0..319
  int b = bc / 10, c = bc % 10;
  int o  = threadIdx.x & 15;
  int rl = threadIdx.x >> 4;    // 0..15
  float acc = 0.0f;
  for (int r = rl; r < 1152; r += 16)
    acc += bij[r * 10 + c] * spike[((b * 1152 + r) * 10 + c) * 16 + o];
  __shared__ float red[256];
  red[threadIdx.x] = acc;
  __syncthreads();
  for (int step = 8; step >= 1; step >>= 1) {
    if (rl < step) red[threadIdx.x] += red[threadIdx.x + step * 16];
    __syncthreads();
  }
  if (rl == 0) sj[bc * 16 + o] = red[o];
}

__global__ void k_dig2(float* __restrict__ mem, float* __restrict__ spike,
                       float* __restrict__ outm, const float* __restrict__ sj) {
  int i = blockIdx.x * blockDim.x + threadIdx.x;
  if (i >= 5120) return;
  float s = sj[i];
  float m = (mem[i] - spike[i] * THRESH) * DECAY + s;
  mem[i] = m;
  spike[i] = m > THRESH ? 1.0f : 0.0f;
  outm[i] += s;
}

// b_ij = clip(b_ij) + 0.0008 * mean_b sum_o (trace-0.1)*d2s ; one block per r
__global__ void k_bij(float* __restrict__ bij, const float* __restrict__ trace,
                      const float* __restrict__ d2s) {
  int r = blockIdx.x;           // 0..1151
  int t = threadIdx.x;          // 0..159
  int c = t >> 4, o = t & 15;
  float acc = 0.0f;
  for (int b = 0; b < 32; ++b)
    acc += (trace[((b * 1152 + r) * 10 + c) * 16 + o] - 0.1f) * d2s[(b * 10 + c) * 16 + o];
  __shared__ float red[160];
  red[t] = acc;
  __syncthreads();
  if (o == 0) {
    float s = 0.0f;
#pragma unroll
    for (int k = 0; k < 16; k++) s += red[c * 16 + k];
    float old = fminf(fmaxf(bij[r * 10 + c], -0.05f), 1.0f);
    bij[r * 10 + c] = old + 0.0008f * (s * (1.0f / 32.0f));
  }
}

__global__ void k_final(const float* __restrict__ outm, float* __restrict__ cls) {
  int t = threadIdx.x;
  if (t >= 320) return;
  float s = 0.0f;
#pragma unroll
  for (int o = 0; o < 16; o++) {
    float v = outm[t * 16 + o] * (1.0f / 5.0f);   // time_window = 5
    s += v * v;
  }
  cls[t] = sqrtf(s);
}

// ---------------- launch ----------------
extern "C" void kernel_launch(void* const* d_in, const int* in_sizes, int n_in,
                              void* d_out, int out_size, void* d_ws, size_t ws_size,
                              hipStream_t stream) {
  const float* data   = (const float*)d_in[0];
  const float* conv_w = (const float*)d_in[1];
  const float* conv_b = (const float*)d_in[2];
  const float* prim_w = (const float*)d_in[3];
  const float* prim_b = (const float*)d_in[4];
  const float* W      = (const float*)d_in[5];
  const float* bias16 = (const float*)d_in[6];
  float* classes = (float*)d_out;               // [32,10]

  char* ws = (char*)d_ws;
  float*    conv_mem  = (float*)(ws + OFF_CONV_MEM);
  float*    prim_mem  = (float*)(ws + OFF_PRIM_MEM);
  float*    prim_spk  = (float*)(ws + OFF_PRIM_SPK);
  float*    dig_mem   = (float*)(ws + OFF_DIG_MEM);
  float*    dig_spk   = (float*)(ws + OFF_DIG_SPK);
  float*    trace_u   = (float*)(ws + OFF_TRACE);
  float*    dig2_mem  = (float*)(ws + OFF_DIG2_MEM);
  float*    dig2_spk  = (float*)(ws + OFF_DIG2_SPK);
  float*    out_mem   = (float*)(ws + OFF_OUT_MEM);
  _Float16* conv_spk  = (_Float16*)(ws + OFF_CONV_SPK);
  float*    bij       = (float*)(ws + OFF_BIJ);
  float*    conv_out  = (float*)(ws + OFF_CONV_OUT);
  _Float16* Bpack     = (_Float16*)(ws + OFF_BPACK);
  _Float16* Apack     = (_Float16*)(ws + OFF_APACK);
  float*    prim_part = (float*)(ws + OFF_PRIM_PART);
  float*    sj        = (float*)(ws + OFF_SJ);

  // re-init all state every call (deterministic across graph replays)
  int zero_words = (int)(ZERO_END / 4u);
  k_zero<<<8192, 256, 0, stream>>>((uint32_t*)ws, zero_words);
  k_init_bij<<<45, 256, 0, stream>>>(bij);
  k_pack_b<<<(16 * KB * 32 + 255) / 256, 256, 0, stream>>>(prim_w, Bpack);
  k_conv1<<<(CONV_N + 255) / 256, 256, 0, stream>>>(data, conv_w, conv_b, conv_out);

  for (int t = 0; t < 5; ++t) {  // time_window = 5
    k_conv_update<<<(CONV_N + 255) / 256, 256, 0, stream>>>(conv_mem, conv_spk, conv_out);
    k_apack<<<(72 * KB * 32 + 255) / 256, 256, 0, stream>>>(conv_spk, Apack);
    k_prim_gemm<<<144, 256, 0, stream>>>(Apack, Bpack, prim_part);
    k_prim_update<<<(PRIM_N + 255) / 256, 256, 0, stream>>>(prim_mem, prim_spk,
                                                            prim_part, prim_b);
    k_dig_update<<<(DIG_N + 255) / 256, 256, 0, stream>>>(W, bias16, prim_spk,
                                                          dig_mem, dig_spk, trace_u);
    k_sj<<<320, 256, 0, stream>>>(bij, dig_spk, sj);
    k_dig2<<<20, 256, 0, stream>>>(dig2_mem, dig2_spk, out_mem, sj);
    k_bij<<<1152, 160, 0, stream>>>(bij, trace_u, dig2_spk);
  }
  k_final<<<1, 320, 0, stream>>>(out_mem, classes);
}
